// ExactRetrieverModule_79233556677244
// MI455X (gfx1250) — matrix operand order repaired and checked by reading, not compiled, not run on hardware
//
#include <hip/hip_runtime.h>
#include <hip/hip_bf16.h>
#include <math.h>

// ---------------------------------------------------------------------------
// Problem constants (from reference)
// ---------------------------------------------------------------------------
#define B_  16
#define S_  2048
#define D_  512
#define N_  200000
#define K_  5
#define NTILES (N_ / 16)      // 12500 doc tiles of 16

typedef __attribute__((ext_vector_type(16))) __bf16 v16bf;
typedef __attribute__((ext_vector_type(8)))  float  v8f;

__device__ __forceinline__ unsigned short f32_to_bf16_rne(float x) {
    unsigned int u = __float_as_uint(x);
    u += 0x7FFFu + ((u >> 16) & 1u);
    return (unsigned short)(u >> 16);
}
__device__ __forceinline__ float bf16u_to_f32(unsigned short h) {
    return __uint_as_float(((unsigned int)h) << 16);
}
// A-matrix (16-bit, 16x32) k-position for VGPR slot v (0..7) and lane half.
__device__ __forceinline__ int a_kloc(int v, int half) {
    return ((v & 4) ? 16 : 0) + 2 * (v & 3) + half * 8;
}

// ---------------------------------------------------------------------------
// Workspace layout (bytes, all 256-aligned).  Total ~13.5 MB.
// ---------------------------------------------------------------------------
#define WS_PARTIAL  0                          // 16*16*512 f32   = 524288
#define WS_POOLED   524288                     // 8192 f32        = 32768
#define WS_QN       557056                     // 8192 f32        = 32768
#define WS_QAHI     589824                     // 512*16 u16      = 16384
#define WS_QALO     606208                     // 512*16 u16      = 16384
#define WS_SCORES   622592                     // 16*200000 f32   = 12800000
#define WS_PARTVAL  13422592                   // 1024*5 f32      = 20480
#define WS_PARTIDX  13443072                   // 1024*5 i32      = 20480
#define WS_TOPVAL   13463552                   // 80 f32 (pad 512)
#define WS_TOPIDX   13464064                   // 80 i32 (pad 512)
#define WS_CTX      13464576                   // 16*512 f32      = 32768

// ---------------------------------------------------------------------------
// 1) Mean pool over S, phase 1: partial sums over 128-row s-chunks
// ---------------------------------------------------------------------------
__global__ void k_pool1(const float* __restrict__ hidden, float* __restrict__ partial) {
    int g  = blockIdx.x * 256 + threadIdx.x;   // 512 blocks * 256 = 131072
    int d  = g & 511;
    int sc = (g >> 9) & 15;
    int b  = g >> 13;
    float acc = 0.f;
    const float* p = hidden + ((size_t)b * S_ + (size_t)sc * 128) * D_ + d;
    #pragma unroll 4
    for (int s = 0; s < 128; ++s) acc += p[(size_t)s * D_];
    partial[((size_t)b * 16 + sc) * D_ + d] = acc;
}

// 2) phase 2: reduce 16 partials -> mean
__global__ void k_pool2(const float* __restrict__ partial, float* __restrict__ pooled) {
    int g = blockIdx.x * 256 + threadIdx.x;    // 32 blocks * 256 = 8192
    int d = g & 511;
    int b = g >> 9;
    float acc = 0.f;
    #pragma unroll
    for (int sc = 0; sc < 16; ++sc) acc += partial[((size_t)b * 16 + sc) * D_ + d];
    pooled[(size_t)b * D_ + d] = acc * (1.0f / (float)S_);
}

// ---------------------------------------------------------------------------
// 3) query = pooled @ W_q^T + b_q, then L2-normalize each row
// ---------------------------------------------------------------------------
__global__ void k_project(const float* __restrict__ pooled,
                          const float* __restrict__ Wq,
                          const float* __restrict__ bq,
                          float* __restrict__ qn) {
    __shared__ float sp[D_];
    __shared__ float red[D_];
    int b = blockIdx.x, tid = threadIdx.x;     // 16 blocks * 512 threads
    sp[tid] = pooled[(size_t)b * D_ + tid];
    __syncthreads();
    float acc = bq[tid];
    const float* wrow = Wq + (size_t)tid * D_;
    #pragma unroll 4
    for (int d = 0; d < D_; ++d) acc += sp[d] * wrow[d];
    red[tid] = acc * acc;
    __syncthreads();
    for (int off = 256; off > 0; off >>= 1) {
        if (tid < off) red[tid] += red[tid + off];
        __syncthreads();
    }
    float inv = 1.0f / fmaxf(sqrtf(red[0]), 1e-12f);
    qn[(size_t)b * D_ + tid] = acc * inv;
}

// ---------------------------------------------------------------------------
// 4) Pre-pack normalized queries into WMMA A-fragment lane layout
//    (16x32 bf16 per k-step t; hi/lo split for f32-accurate emulation)
// ---------------------------------------------------------------------------
__global__ void k_qpack(const float* __restrict__ qn,
                        unsigned short* __restrict__ qAhi,
                        unsigned short* __restrict__ qAlo) {
    int tid  = threadIdx.x;                    // 1 block * 512: tid = t*32 + lane
    int t    = tid >> 5;
    int lane = tid & 31;
    int m    = lane & 15;
    int half = lane >> 4;
    #pragma unroll
    for (int v = 0; v < 8; ++v) {
        int k = t * 32 + a_kloc(v, half);
        #pragma unroll
        for (int e = 0; e < 2; ++e) {
            float x = qn[(size_t)m * D_ + k + e];
            unsigned short hi = f32_to_bf16_rne(x);
            unsigned short lo = f32_to_bf16_rne(x - bf16u_to_f32(hi));
            qAhi[(size_t)tid * 16 + 2 * v + e] = hi;
            qAlo[(size_t)tid * 16 + 2 * v + e] = lo;
        }
    }
}

// ---------------------------------------------------------------------------
// 5) Cosine scan: scores[16, N] = qn @ docsN^T, fusing per-doc L2 norm.
//    One wave per 16-doc tile.  Split-bf16 (hi/lo) with THREE independent
//    accumulator chains so the 3 WMMAs per K-step issue back-to-back.
//    K loop kept rolled (#pragma unroll 1) to avoid scratch spills.
// ---------------------------------------------------------------------------
__global__ void __launch_bounds__(256) k_scan(const float* __restrict__ docs,
                       const unsigned short* __restrict__ qAhi,
                       const unsigned short* __restrict__ qAlo,
                       float* __restrict__ scores) {
    const int lane  = threadIdx.x & 31;
    const int half  = lane >> 4;
    const int nn    = lane & 15;
    const int wave  = (blockIdx.x * (blockDim.x >> 5)) + (threadIdx.x >> 5);
    const int nwave = gridDim.x * (blockDim.x >> 5);
    const v16bf* aHi = reinterpret_cast<const v16bf*>(qAhi);
    const v16bf* aLo = reinterpret_cast<const v16bf*>(qAlo);

    #pragma unroll 1
    for (int tile = wave; tile < NTILES; tile += nwave) {
        const float* drow = docs + ((size_t)tile * 16 + nn) * D_;
        if (tile + nwave < NTILES)   // wave-uniform: hint next tile into cache
            __builtin_prefetch(docs + ((size_t)(tile + nwave) * 16 + nn) * D_, 0, 1);

        v8f ahh = {}, ahl = {}, alh = {};
        float n2 = 0.f;
        #pragma unroll 1
        for (int t = 0; t < 16; ++t) {
            v16bf ah = aHi[t * 32 + lane];
            v16bf al = aLo[t * 32 + lane];
            v16bf bh, bl;
            const float* src = drow + t * 32;
            #pragma unroll
            for (int v = 0; v < 8; ++v) {
                float2 x = *(const float2*)(src + a_kloc(v, half));
                n2 = fmaf(x.x, x.x, fmaf(x.y, x.y, n2));
                __bf16 h0 = (__bf16)x.x;       // native v_cvt (RNE)
                __bf16 h1 = (__bf16)x.y;
                bh[2 * v]     = h0;
                bh[2 * v + 1] = h1;
                bl[2 * v]     = (__bf16)(x.x - (float)h0);
                bl[2 * v + 1] = (__bf16)(x.y - (float)h1);
            }
            ahh = __builtin_amdgcn_wmma_f32_16x16x32_bf16(false, ah, false, bh,
                                                          (short)0, ahh, false, false);
            ahl = __builtin_amdgcn_wmma_f32_16x16x32_bf16(false, ah, false, bl,
                                                          (short)0, ahl, false, false);
            alh = __builtin_amdgcn_wmma_f32_16x16x32_bf16(false, al, false, bh,
                                                          (short)0, alh, false, false);
        }
        v8f acc = ahh + ahl + alh;
        // combine the two K-half partial norms for doc nn (lanes n and n+16)
        n2 += __shfl_xor(n2, 16, 32);
        float inv = 1.0f / fmaxf(sqrtf(n2), 1e-12f);
        #pragma unroll
        for (int r = 0; r < 8; ++r) {
            int m = r + half * 8;
            scores[(size_t)m * N_ + (size_t)tile * 16 + nn] = acc[r] * inv;
        }
    }
}

// ---------------------------------------------------------------------------
// 6) Top-K phase 1: per (b, chunk) block keeps top-5 of its 3125 scores
// ---------------------------------------------------------------------------
__global__ void k_topk1(const float* __restrict__ scores,
                        float* __restrict__ partVal, int* __restrict__ partIdx) {
    __shared__ float rv[256];
    __shared__ int   ri[256];
    __shared__ int   rs[256];
    __shared__ int   sWin;
    int tid = threadIdx.x;
    int b     = blockIdx.x >> 6;               // 1024 blocks = 16 b * 64 chunks
    int chunk = blockIdx.x & 63;
    int n0    = chunk * (N_ / 64);             // 3125 per chunk
    const float* row = scores + (size_t)b * N_ + n0;

    float cs[K_]; int ci[K_];
    #pragma unroll
    for (int k = 0; k < K_; ++k) { cs[k] = -3.0e38f; ci[k] = -1; }
    for (int n = tid; n < N_ / 64; n += 256) {
        float v = row[n];
        if (v > cs[K_ - 1]) {
            int j = K_ - 1;
            while (j > 0 && v > cs[j - 1]) { cs[j] = cs[j - 1]; ci[j] = ci[j - 1]; --j; }
            cs[j] = v; ci[j] = n0 + n;
        }
    }
    int pos = 0;
    for (int k = 0; k < K_; ++k) {
        rv[tid] = (pos < K_) ? cs[pos] : -3.0e38f;
        ri[tid] = (pos < K_) ? ci[pos] : -1;
        rs[tid] = tid;
        __syncthreads();
        for (int off = 128; off > 0; off >>= 1) {
            if (tid < off && rv[tid + off] > rv[tid]) {
                rv[tid] = rv[tid + off]; ri[tid] = ri[tid + off]; rs[tid] = rs[tid + off];
            }
            __syncthreads();
        }
        if (tid == 0) {
            partVal[blockIdx.x * K_ + k] = rv[0];
            partIdx[blockIdx.x * K_ + k] = ri[0];
            sWin = rs[0];
        }
        __syncthreads();
        if (tid == sWin) ++pos;
        __syncthreads();
    }
}

// 7) Top-K phase 2: merge 64*5 candidates per b, emit final top-5 + indices
__global__ void k_topk2(const float* __restrict__ partVal, const int* __restrict__ partIdx,
                        float* __restrict__ topVal, int* __restrict__ topIdx,
                        float* __restrict__ out) {
    __shared__ float mv[512];
    __shared__ int   mi[512];
    __shared__ float rv[512];
    __shared__ int   rs[512];
    int tid = threadIdx.x, b = blockIdx.x;     // 16 blocks * 512
    const int NC = 64 * K_;                    // 320 candidates
    mv[tid] = (tid < NC) ? partVal[(size_t)b * NC + tid] : -3.0e38f;
    mi[tid] = (tid < NC) ? partIdx[(size_t)b * NC + tid] : 0;
    __syncthreads();
    for (int k = 0; k < K_; ++k) {
        rv[tid] = mv[tid]; rs[tid] = tid;
        __syncthreads();
        for (int off = 256; off > 0; off >>= 1) {
            if (tid < off && rv[tid + off] > rv[tid]) {
                rv[tid] = rv[tid + off]; rs[tid] = rs[tid + off];
            }
            __syncthreads();
        }
        if (tid == 0) {
            int w = rs[0]; float v = rv[0]; int di = mi[w];
            topVal[b * K_ + k] = v;
            topIdx[b * K_ + k] = di;
            out[b * K_ + k]            = v;          // top_scores
            out[B_ * K_ + b * K_ + k]  = (float)di;  // indices (as f32 values)
            mv[w] = -3.0e38f;
        }
        __syncthreads();
    }
}

// ---------------------------------------------------------------------------
// 8) Context: softmax(top-5) weighted sum of normalized retrieved docs
// ---------------------------------------------------------------------------
__global__ void k_context(const float* __restrict__ docs,
                          const float* __restrict__ topVal, const int* __restrict__ topIdx,
                          float* __restrict__ ctx) {
    __shared__ float red[512];
    __shared__ float sInv[K_];
    __shared__ float sW[K_];
    int tid = threadIdx.x, b = blockIdx.x;     // 16 blocks * 512
    for (int k = 0; k < K_; ++k) {
        const float* row = docs + (size_t)topIdx[b * K_ + k] * D_;
        float x = row[tid];
        red[tid] = x * x;
        __syncthreads();
        for (int off = 256; off > 0; off >>= 1) {
            if (tid < off) red[tid] += red[tid + off];
            __syncthreads();
        }
        if (tid == 0) sInv[k] = 1.0f / fmaxf(sqrtf(red[0]), 1e-12f);
        __syncthreads();
    }
    if (tid == 0) {
        float s[K_], mx = -3.0e38f, sum = 0.f;
        #pragma unroll
        for (int k = 0; k < K_; ++k) { s[k] = topVal[b * K_ + k]; mx = fmaxf(mx, s[k]); }
        #pragma unroll
        for (int k = 0; k < K_; ++k) { s[k] = __expf(s[k] - mx); sum += s[k]; }
        #pragma unroll
        for (int k = 0; k < K_; ++k) sW[k] = s[k] / sum;
    }
    __syncthreads();
    float acc = 0.f;
    #pragma unroll
    for (int k = 0; k < K_; ++k)
        acc += sW[k] * sInv[k] * docs[(size_t)topIdx[b * K_ + k] * D_ + tid];
    ctx[(size_t)b * D_ + tid] = acc;
}

// ---------------------------------------------------------------------------
// 9) Gate GEMM + fuse: for each 16-row tile, WMMA over K=1024 (hidden|ctx),
//    sigmoid, blend.  8 waves split 512 output channels; A staged in LDS.
//    4 nt-tiles give 4 independent accumulator chains; t loop kept rolled.
// ---------------------------------------------------------------------------
__global__ void __launch_bounds__(256) k_gate(const float* __restrict__ hidden,
                                              const float* __restrict__ Wg,
                                              const float* __restrict__ bg,
                                              const float* __restrict__ ctx,
                                              float* __restrict__ out) {
    __shared__ float sA[16 * 132];   // 16 rows x 128 k, padded (stride 132)
    __shared__ float sCtx[D_];
    const int tid  = threadIdx.x;
    const int b    = blockIdx.x >> 7;          // 2048 blocks = 16 b * 128 s-tiles
    const int s0   = (blockIdx.x & 127) * 16;
    const int wave = tid >> 5;
    const int lane = tid & 31;
    const int half = lane >> 4;
    const int nn   = lane & 15;
    const int mrow = lane & 15;

    for (int i = tid; i < D_; i += 256) sCtx[i] = ctx[(size_t)b * D_ + i];

    v8f acc0 = {}, acc1 = {}, acc2 = {}, acc3 = {};
    #pragma unroll 1
    for (int c = 0; c < 8; ++c) {              // 8 k-chunks of 128
        __syncthreads();
        for (int idx = tid; idx < 16 * 128; idx += 256) {
            int row = idx >> 7, kk = idx & 127;
            float v;
            if (c < 4)
                v = hidden[((size_t)(b * S_ + s0 + row)) * D_ + c * 128 + kk];
            else
                v = sCtx[(c - 4) * 128 + kk];
            sA[row * 132 + kk] = v;
        }
        __syncthreads();
        #pragma unroll 1
        for (int t = 0; t < 4; ++t) {
            v16bf ah, al;
            #pragma unroll
            for (int v = 0; v < 8; ++v) {
                int kin = t * 32 + a_kloc(v, half);
                float2 x = *(const float2*)&sA[mrow * 132 + kin];
                __bf16 h0 = (__bf16)x.x;
                __bf16 h1 = (__bf16)x.y;
                ah[2 * v]     = h0;
                ah[2 * v + 1] = h1;
                al[2 * v]     = (__bf16)(x.x - (float)h0);
                al[2 * v + 1] = (__bf16)(x.y - (float)h1);
            }
            #pragma unroll
            for (int nt = 0; nt < 4; ++nt) {
                int i = wave * 64 + nt * 16 + nn;
                const float* wrow = Wg + (size_t)i * (2 * D_) + c * 128 + t * 32;
                v16bf bh, bl;
                #pragma unroll
                for (int v = 0; v < 8; ++v) {
                    float2 x = *(const float2*)(wrow + a_kloc(v, half));
                    __bf16 h0 = (__bf16)x.x;
                    __bf16 h1 = (__bf16)x.y;
                    bh[2 * v]     = h0;
                    bh[2 * v + 1] = h1;
                    bl[2 * v]     = (__bf16)(x.x - (float)h0);
                    bl[2 * v + 1] = (__bf16)(x.y - (float)h1);
                }
                v8f a = (nt == 0) ? acc0 : (nt == 1) ? acc1 : (nt == 2) ? acc2 : acc3;
                a = __builtin_amdgcn_wmma_f32_16x16x32_bf16(false, al, false, bh,
                                                            (short)0, a, false, false);
                a = __builtin_amdgcn_wmma_f32_16x16x32_bf16(false, ah, false, bl,
                                                            (short)0, a, false, false);
                a = __builtin_amdgcn_wmma_f32_16x16x32_bf16(false, ah, false, bh,
                                                            (short)0, a, false, false);
                if (nt == 0) acc0 = a; else if (nt == 1) acc1 = a;
                else if (nt == 2) acc2 = a; else acc3 = a;
            }
        }
    }
    // epilogue: sigmoid gate + blend, write fused at out[160 + ...]
    const float* hBase = hidden + (size_t)(b * S_ + s0) * D_;
    float* oBase = out + 2 * B_ * K_ + (size_t)(b * S_ + s0) * D_;
    #pragma unroll
    for (int nt = 0; nt < 4; ++nt) {
        int i = wave * 64 + nt * 16 + nn;
        float bgi = bg[i];
        float cv  = sCtx[i];
        v8f a = (nt == 0) ? acc0 : (nt == 1) ? acc1 : (nt == 2) ? acc2 : acc3;
        #pragma unroll
        for (int r = 0; r < 8; ++r) {
            int m = r + half * 8;
            float g = 1.0f / (1.0f + __expf(-(a[r] + bgi)));
            float h = hBase[(size_t)m * D_ + i];
            oBase[(size_t)m * D_ + i] = g * h + (1.0f - g) * cv;
        }
    }
}

// ---------------------------------------------------------------------------
// Launch
// ---------------------------------------------------------------------------
extern "C" void kernel_launch(void* const* d_in, const int* in_sizes, int n_in,
                              void* d_out, int out_size, void* d_ws, size_t ws_size,
                              hipStream_t stream) {
    const float* hidden = (const float*)d_in[0];
    const float* docs   = (const float*)d_in[1];
    const float* Wq     = (const float*)d_in[2];
    const float* bq     = (const float*)d_in[3];
    const float* Wg     = (const float*)d_in[4];
    const float* bgate  = (const float*)d_in[5];
    float* out = (float*)d_out;

    char* w = (char*)d_ws;
    float*          wsPartial = (float*)(w + WS_PARTIAL);
    float*          wsPooled  = (float*)(w + WS_POOLED);
    float*          wsQn      = (float*)(w + WS_QN);
    unsigned short* wsQAhi    = (unsigned short*)(w + WS_QAHI);
    unsigned short* wsQAlo    = (unsigned short*)(w + WS_QALO);
    float*          wsScores  = (float*)(w + WS_SCORES);
    float*          wsPartVal = (float*)(w + WS_PARTVAL);
    int*            wsPartIdx = (int*)(w + WS_PARTIDX);
    float*          wsTopVal  = (float*)(w + WS_TOPVAL);
    int*            wsTopIdx  = (int*)(w + WS_TOPIDX);
    float*          wsCtx     = (float*)(w + WS_CTX);

    k_pool1  <<<512,  256, 0, stream>>>(hidden, wsPartial);
    k_pool2  <<<32,   256, 0, stream>>>(wsPartial, wsPooled);
    k_project<<<16,   512, 0, stream>>>(wsPooled, Wq, bq, wsQn);
    k_qpack  <<<1,    512, 0, stream>>>(wsQn, wsQAhi, wsQAlo);
    k_scan   <<<640,  256, 0, stream>>>(docs, wsQAhi, wsQAlo, wsScores);
    k_topk1  <<<1024, 256, 0, stream>>>(wsScores, wsPartVal, wsPartIdx);
    k_topk2  <<<16,   512, 0, stream>>>(wsPartVal, wsPartIdx, wsTopVal, wsTopIdx, out);
    k_context<<<16,   512, 0, stream>>>(docs, wsTopVal, wsTopIdx, wsCtx);
    k_gate   <<<2048, 256, 0, stream>>>(hidden, Wg, bgate, wsCtx, out);
}
